// GraphConvolutionLayer_4045859193005
// MI455X (gfx1250) — matrix operand order repaired and checked by reading, not compiled, not run on hardware
//
#include <hip/hip_runtime.h>
#include <stdint.h>

// ---------------------------------------------------------------------------
// CDNA5 WMMA (wave32): D(16x16 f32) = A(16x32 bf16) x B(32x16 bf16) + C
// All GEMM operands are pre-converted to bf16; fragments are fetched straight
// from memory with per-lane contiguous b128 loads (ISA 10.9 row-major rule).
// ---------------------------------------------------------------------------
typedef __attribute__((ext_vector_type(16))) __bf16 bf16x16;
typedef __attribute__((ext_vector_type(8)))  float  f32x8;

struct Frag {
    union { uint32_t u[8]; bf16x16 v; };
};

__device__ __forceinline__ f32x8 zero8() {
    f32x8 z;
#pragma unroll
    for (int i = 0; i < 8; ++i) z[i] = 0.0f;
    return z;
}

// ---- f32 -> packed 2x bf16 (round-to-nearest) ------------------------------
#if __has_builtin(__builtin_amdgcn_cvt_pk_bf16_f32)
__device__ __forceinline__ uint32_t pk2(float lo, float hi) {
    auto r = __builtin_amdgcn_cvt_pk_bf16_f32(lo, hi);
    return __builtin_bit_cast(uint32_t, r);
}
#else
__device__ __forceinline__ uint32_t pk2(float lo, float hi) {
    union { float f; uint32_t u; } a, b; a.f = lo; b.f = hi;
    return ((a.u + 0x8000u) >> 16) | ((b.u + 0x8000u) & 0xFFFF0000u);
}
#endif
__device__ __forceinline__ uint16_t f2bf1(float x) {
    union { float f; uint32_t u; } c; c.f = x;
    return (uint16_t)((c.u + 0x8000u) >> 16);
}

// ---- direct bf16 fragment loads -------------------------------------------
// A fragment; src points at (row0, k0/2) in dwords, stride in dwords.
__device__ __forceinline__ Frag load_a_bf16(const uint32_t* src, int rowStrideU) {
    const int L = threadIdx.x & 31, M = L & 15, g = L >> 4;
    const uint32_t* p = src + (size_t)M * rowStrideU;
    uint4 x = *reinterpret_cast<const uint4*>(p + g * 4);       // K = g*8..+7
    uint4 y = *reinterpret_cast<const uint4*>(p + 8 + g * 4);   // K = 16+g*8..
    Frag f;
    f.u[0] = x.x; f.u[1] = x.y; f.u[2] = x.z; f.u[3] = x.w;
    f.u[4] = y.x; f.u[5] = y.y; f.u[6] = y.z; f.u[7] = y.w;
    return f;
}
// B fragment where B[k][n] = srcRow(n) contiguous in k; src at (n0, k0/2).
__device__ __forceinline__ Frag load_bT_bf16(const uint32_t* src, int rowStrideU) {
    const int L = threadIdx.x & 31, N = L & 15, g = L >> 4;
    const uint32_t* p = src + (size_t)N * rowStrideU;           // K = g*8..+7, 16+g*8..+7
    uint4 x = *reinterpret_cast<const uint4*>(p + g * 4);
    uint4 y = *reinterpret_cast<const uint4*>(p + 8 + g * 4);
    Frag f;
    f.u[0] = x.x; f.u[1] = x.y; f.u[2] = x.z; f.u[3] = x.w;
    f.u[4] = y.x; f.u[5] = y.y; f.u[6] = y.z; f.u[7] = y.w;
    return f;
}

__device__ __forceinline__ f32x8 wmma_bf16(Frag a, Frag b, f32x8 c) {
    return __builtin_amdgcn_wmma_f32_16x16x32_bf16(false, a.v, false, b.v, (short)0, c, false, false);
}
__device__ __forceinline__ float sigmoidf_(float x) { return 1.0f / (1.0f + __expf(-x)); }

// ---------------------------------------------------------------------------
#define NN 512
#define TT 32
#define DD 512
#define RR 3
#define TD  (TT * DD)        // 16384 elems
#define TDU (TD / 2)         // row stride of featsB in dwords
#define DDU (DD / 2)         // 256 dwords
#define D3  (3 * DD)         // 1536

// ---------------------------------------------------------------------------
// Flat f32 -> bf16 convert (8 floats / thread)
// ---------------------------------------------------------------------------
__global__ __launch_bounds__(256) void k_cvt(const float* __restrict__ src,
                                             uint32_t* __restrict__ dst, int n4) {
    int i = blockIdx.x * blockDim.x + threadIdx.x;
    if (i >= n4) return;
    const float4* s = reinterpret_cast<const float4*>(src) + (size_t)i * 2;
    float4 a = s[0], b = s[1];
    uint4 o;
    o.x = pk2(a.x, a.y); o.y = pk2(a.z, a.w);
    o.z = pk2(b.x, b.y); o.w = pk2(b.z, b.w);
    reinterpret_cast<uint4*>(dst)[i] = o;
}

// ---------------------------------------------------------------------------
// adj f32 -> bf16 + fused row-sum (one wave per 512-wide row)
// ---------------------------------------------------------------------------
__global__ __launch_bounds__(256) void k_cvt_adj(const float* __restrict__ adj,
                                                 uint32_t* __restrict__ adjB,
                                                 float* __restrict__ rs) {
    const int wid = threadIdx.x >> 5, lane = threadIdx.x & 31;
    const size_t row = (size_t)blockIdx.x * 8 + wid;
    const float4* p = reinterpret_cast<const float4*>(adj + row * NN + lane * 16);
    float4 x0 = p[0], x1 = p[1], x2 = p[2], x3 = p[3];
    float s = x0.x + x0.y + x0.z + x0.w + x1.x + x1.y + x1.z + x1.w +
              x2.x + x2.y + x2.z + x2.w + x3.x + x3.y + x3.z + x3.w;
    uint4 o0, o1;
    o0.x = pk2(x0.x, x0.y); o0.y = pk2(x0.z, x0.w);
    o0.z = pk2(x1.x, x1.y); o0.w = pk2(x1.z, x1.w);
    o1.x = pk2(x2.x, x2.y); o1.y = pk2(x2.z, x2.w);
    o1.z = pk2(x3.x, x3.y); o1.w = pk2(x3.z, x3.w);
    uint4* d = reinterpret_cast<uint4*>(adjB + row * DDU + lane * 8);
    d[0] = o0; d[1] = o1;
    for (int off = 16; off > 0; off >>= 1) s += __shfl_xor(s, off, 32);
    if (lane == 0) rs[row] = s;
}

// ---------------------------------------------------------------------------
// Transpose f32 [rows x 512] -> bf16 dst[512 x 512] (dst[e][m] = src[m][e])
// ---------------------------------------------------------------------------
__global__ __launch_bounds__(256) void k_transpose(const float* __restrict__ srcBase,
                                                   long srcBatch, int srcRowStride,
                                                   uint16_t* __restrict__ dstBase,
                                                   long dstBatch) {
    __shared__ float tile[32][33];
    const int e0 = blockIdx.x * 32, m0 = blockIdx.y * 32, b = blockIdx.z;
    const float* src = srcBase + (size_t)b * srcBatch;
    uint16_t* dst = dstBase + (size_t)b * dstBatch;
#pragma unroll
    for (int q = 0; q < 4; ++q) {
        int m = threadIdx.y + 8 * q;
        tile[m][threadIdx.x] = src[(size_t)(m0 + m) * srcRowStride + e0 + threadIdx.x];
    }
    __syncthreads();
#pragma unroll
    for (int q = 0; q < 4; ++q) {
        int e = threadIdx.y + 8 * q;
        dst[(size_t)(e0 + e) * 512 + m0 + threadIdx.x] = f2bf1(tile[threadIdx.x][e]);
    }
}

// ---------------------------------------------------------------------------
// WcombT[r][j][d] = sum_e w_ih[j,e] * W_r[d,e]   (bf16 in / bf16 out)
// ---------------------------------------------------------------------------
__global__ __launch_bounds__(32) void k_wcomb(const uint32_t* __restrict__ w_ihB,
                                              const uint32_t* __restrict__ weightB,
                                              uint16_t* __restrict__ wcombT) {
    const int d0 = blockIdx.x * 16, j0 = blockIdx.y * 16, r = blockIdx.z;
    const uint32_t* wr = weightB + (size_t)r * DD * DDU;
    f32x8 c = zero8();
    for (int k0 = 0; k0 < DD; k0 += 32) {
        Frag a = load_a_bf16(w_ihB + (size_t)j0 * DDU + (k0 >> 1), DDU);
        Frag b = load_bT_bf16(wr + (size_t)d0 * DDU + (k0 >> 1), DDU);
        c = wmma_bf16(a, b, c);
    }
    const int L = threadIdx.x & 31, col = L & 15, g = L >> 4;
#pragma unroll
    for (int j = 0; j < 8; ++j)
        wcombT[((size_t)r * D3 + j0 + j + 8 * g) * DD + d0 + col] = f2bf1(c[j]);
}

// ---------------------------------------------------------------------------
// bvec[r][j] = sum_e bias[r,0,e] * w_ih[j,e]
// ---------------------------------------------------------------------------
__global__ __launch_bounds__(256) void k_bvec(const float* __restrict__ bias,
                                              const float* __restrict__ w_ih,
                                              float* __restrict__ bvec) {
    int idx = blockIdx.x * blockDim.x + threadIdx.x;
    if (idx >= RR * D3) return;
    int r = idx / D3, j = idx % D3;
    const float* b = bias + (size_t)r * DD;
    const float* w = w_ih + (size_t)j * DD;
    float s = 0.0f;
    for (int e = 0; e < DD; ++e) s += b[e] * w[e];
    bvec[idx] = s;
}

// ---------------------------------------------------------------------------
// P[rt][n][e] = sum_m adjB[rt][n][m] * FT[t][e][m]   (32x32 tile per wave)
// ---------------------------------------------------------------------------
__global__ __launch_bounds__(32) void k_pmm(const uint32_t* __restrict__ adjB,
                                            const uint32_t* __restrict__ FT,
                                            uint16_t* __restrict__ P) {
    const int e0 = blockIdx.x * 32, n0 = blockIdx.y * 32, rt = blockIdx.z;
    const int t = rt & (TT - 1);
    const uint32_t* A = adjB + (size_t)rt * NN * DDU + (size_t)n0 * DDU;
    const uint32_t* B = FT + (size_t)t * DD * DDU + (size_t)e0 * DDU;
    f32x8 c00 = zero8(), c01 = zero8(), c10 = zero8(), c11 = zero8();
    for (int k0 = 0; k0 < NN; k0 += 32) {
        Frag a0 = load_a_bf16(A + (k0 >> 1), DDU);
        Frag a1 = load_a_bf16(A + (size_t)16 * DDU + (k0 >> 1), DDU);
        Frag b0 = load_bT_bf16(B + (k0 >> 1), DDU);
        Frag b1 = load_bT_bf16(B + (size_t)16 * DDU + (k0 >> 1), DDU);
        c00 = wmma_bf16(a0, b0, c00);
        c01 = wmma_bf16(a0, b1, c01);
        c10 = wmma_bf16(a1, b0, c10);
        c11 = wmma_bf16(a1, b1, c11);
    }
    const int L = threadIdx.x & 31, col = L & 15, g = L >> 4;
    uint16_t* out = P + (size_t)rt * NN * DD;
#pragma unroll
    for (int j = 0; j < 8; ++j) {
        size_t row0 = (size_t)(n0 + j + 8 * g), row1 = row0 + 16;
        out[row0 * DD + e0 + col]      = f2bf1(c00[j]);
        out[row0 * DD + e0 + 16 + col] = f2bf1(c01[j]);
        out[row1 * DD + e0 + col]      = f2bf1(c10[j]);
        out[row1 * DD + e0 + 16 + col] = f2bf1(c11[j]);
    }
}

// ---------------------------------------------------------------------------
// Fused GRU + highway: 32 rows x 16 cols per wave, all-bf16 operands.
// ---------------------------------------------------------------------------
__global__ __launch_bounds__(32) void k_fused(const float* __restrict__ inputs,
                                              const uint32_t* __restrict__ featsB,
                                              const uint32_t* __restrict__ P,
                                              const uint32_t* __restrict__ wcombT,
                                              const uint32_t* __restrict__ whwT,
                                              const uint32_t* __restrict__ w_hhB,
                                              const float* __restrict__ bvec,
                                              const float* __restrict__ rs,
                                              const float* __restrict__ b_ih,
                                              const float* __restrict__ b_hh,
                                              const float* __restrict__ bhw,
                                              float* __restrict__ out) {
    const int e0 = blockIdx.x * 16, n0 = blockIdx.y * 32, t = blockIdx.z;

    // Phase 1: gh gates = feats_t @ w_hh^T, tac = feats_t @ whw   (2 row-halves)
    f32x8 ghr[2], ghz[2], ghn[2], tac[2];
#pragma unroll
    for (int h = 0; h < 2; ++h) { ghr[h] = zero8(); ghz[h] = zero8(); ghn[h] = zero8(); tac[h] = zero8(); }
    const uint32_t* Af = featsB + (size_t)n0 * TDU + (size_t)t * DDU;
    for (int k0 = 0; k0 < DD; k0 += 32) {
        Frag a0 = load_a_bf16(Af + (k0 >> 1), TDU);
        Frag a1 = load_a_bf16(Af + (size_t)16 * TDU + (k0 >> 1), TDU);
        Frag br = load_bT_bf16(w_hhB + (size_t)(0 * DD + e0) * DDU + (k0 >> 1), DDU);
        Frag bz = load_bT_bf16(w_hhB + (size_t)(1 * DD + e0) * DDU + (k0 >> 1), DDU);
        Frag bn = load_bT_bf16(w_hhB + (size_t)(2 * DD + e0) * DDU + (k0 >> 1), DDU);
        Frag bw = load_bT_bf16(whwT + (size_t)e0 * DDU + (k0 >> 1), DDU);
        ghr[0] = wmma_bf16(a0, br, ghr[0]);  ghr[1] = wmma_bf16(a1, br, ghr[1]);
        ghz[0] = wmma_bf16(a0, bz, ghz[0]);  ghz[1] = wmma_bf16(a1, bz, ghz[1]);
        ghn[0] = wmma_bf16(a0, bn, ghn[0]);  ghn[1] = wmma_bf16(a1, bn, ghn[1]);
        tac[0] = wmma_bf16(a0, bw, tac[0]);  tac[1] = wmma_bf16(a1, bw, tac[1]);
    }

    const int L = threadIdx.x & 31, col = L & 15, g = L >> 4;
    const int e = e0 + col;
    const float bhhr = b_hh[e], bhhz = b_hh[DD + e], bhhn = b_hh[2 * DD + e];
    const float bihr = b_ih[e], bihz = b_ih[DD + e], bihn = b_ih[2 * DD + e];
    const float bh = bhw[e];
    float hval[2][8];
#pragma unroll
    for (int h = 0; h < 2; ++h)
#pragma unroll
        for (int j = 0; j < 8; ++j)
            hval[h][j] = inputs[(size_t)(n0 + 16 * h + j + 8 * g) * TD + (size_t)t * DD + e];

    // Phase 2: per relation gi gates + GRU accumulate
    f32x8 oac[2] = {zero8(), zero8()};
    for (int r = 0; r < RR; ++r) {
        const int rt = r * TT + t;
        f32x8 gir[2], giz[2], gin[2];
#pragma unroll
        for (int h = 0; h < 2; ++h) { gir[h] = zero8(); giz[h] = zero8(); gin[h] = zero8(); }
        const uint32_t* Ap = P + ((size_t)rt * NN + n0) * DDU;
        const uint32_t* B0 = wcombT + ((size_t)r * D3 + e0) * DDU;
        for (int k0 = 0; k0 < DD; k0 += 32) {
            Frag a0 = load_a_bf16(Ap + (k0 >> 1), DDU);
            Frag a1 = load_a_bf16(Ap + (size_t)16 * DDU + (k0 >> 1), DDU);
            Frag br = load_bT_bf16(B0 + (k0 >> 1), DDU);
            Frag bz = load_bT_bf16(B0 + (size_t)DD * DDU + (k0 >> 1), DDU);
            Frag bn = load_bT_bf16(B0 + (size_t)(2 * DD) * DDU + (k0 >> 1), DDU);
            gir[0] = wmma_bf16(a0, br, gir[0]);  gir[1] = wmma_bf16(a1, br, gir[1]);
            giz[0] = wmma_bf16(a0, bz, giz[0]);  giz[1] = wmma_bf16(a1, bz, giz[1]);
            gin[0] = wmma_bf16(a0, bn, gin[0]);  gin[1] = wmma_bf16(a1, bn, gin[1]);
        }
        const float bvr = bvec[r * D3 + e];
        const float bvz = bvec[r * D3 + DD + e];
        const float bvn = bvec[r * D3 + 2 * DD + e];
#pragma unroll
        for (int h = 0; h < 2; ++h) {
#pragma unroll
            for (int j = 0; j < 8; ++j) {
                float rsv = rs[(size_t)rt * NN + n0 + 16 * h + j + 8 * g];
                float ir  = gir[h][j] + rsv * bvr + bihr;
                float iz  = giz[h][j] + rsv * bvz + bihz;
                float in_ = gin[h][j] + rsv * bvn + bihn;
                float rg = sigmoidf_(ir + ghr[h][j] + bhhr);
                float zg = sigmoidf_(iz + ghz[h][j] + bhhz);
                float ng = tanhf(in_ + rg * (ghn[h][j] + bhhn));
                oac[h][j] += (1.0f - zg) * ng + zg * hval[h][j];
            }
        }
    }

    // Epilogue: relu, highway blend, store [N, T, E]
#pragma unroll
    for (int h = 0; h < 2; ++h)
#pragma unroll
        for (int j = 0; j < 8; ++j) {
            float o = oac[h][j] > 0.0f ? oac[h][j] : 0.0f;
            float tr = sigmoidf_(tac[h][j] + bh);
            out[(size_t)(n0 + 16 * h + j + 8 * g) * TD + (size_t)t * DD + e] =
                o * tr + hval[h][j] * (1.0f - tr);
        }
}

// ---------------------------------------------------------------------------
// Launcher
// ---------------------------------------------------------------------------
extern "C" void kernel_launch(void* const* d_in, const int* in_sizes, int n_in,
                              void* d_out, int out_size, void* d_ws, size_t ws_size,
                              hipStream_t stream) {
    const float* inputs = (const float*)d_in[0];   // [N,T,D]
    const float* adj    = (const float*)d_in[1];   // [R,T,N,N]
    const float* weight = (const float*)d_in[2];   // [R,D,D]
    const float* bias   = (const float*)d_in[3];   // [R,1,D]
    const float* w_ih   = (const float*)d_in[4];   // [3D,D]
    const float* w_hh   = (const float*)d_in[5];   // [3D,D]
    const float* b_ih   = (const float*)d_in[6];   // [3D]
    const float* b_hh   = (const float*)d_in[7];   // [3D]
    const float* whw    = (const float*)d_in[8];   // [D,D]
    const float* bhw    = (const float*)d_in[9];   // [1,D]
    float* outp = (float*)d_out;

    // Workspace carve-up (all offsets 16B aligned)
    char* w = (char*)d_ws;
    size_t off = 0;
    auto carve = [&](size_t bytes) { void* p = w + off; off += bytes; return p; };
    uint16_t* Pb      = (uint16_t*)carve((size_t)RR * TT * NN * DD * 2);  // 50.3 MB
    uint16_t* wcombT  = (uint16_t*)carve((size_t)RR * D3 * DD * 2);       //  4.7 MB
    float*    bvec    = (float*)   carve((size_t)RR * D3 * 4);
    float*    rs      = (float*)   carve((size_t)RR * TT * NN * 4);
    uint16_t* FT      = (uint16_t*)carve((size_t)TT * DD * NN * 2);       // 16.8 MB
    uint16_t* whwT    = (uint16_t*)carve((size_t)DD * DD * 2);
    uint16_t* adjB    = (uint16_t*)carve((size_t)RR * TT * NN * NN * 2);  // 50.3 MB
    uint16_t* featsB  = (uint16_t*)carve((size_t)NN * TT * DD * 2);       // 16.8 MB
    uint16_t* w_ihB   = (uint16_t*)carve((size_t)D3 * DD * 2);
    uint16_t* w_hhB   = (uint16_t*)carve((size_t)D3 * DD * 2);
    uint16_t* weightB = (uint16_t*)carve((size_t)RR * DD * DD * 2);

    // One-time conversions (bandwidth-bound; pk2 VALU paid exactly once/elem)
    k_cvt<<<dim3((NN * TT * DD / 8 + 255) / 256), 256, 0, stream>>>(inputs, (uint32_t*)featsB, NN * TT * DD / 8);
    k_cvt<<<dim3((D3 * DD / 8 + 255) / 256), 256, 0, stream>>>(w_ih, (uint32_t*)w_ihB, D3 * DD / 8);
    k_cvt<<<dim3((D3 * DD / 8 + 255) / 256), 256, 0, stream>>>(w_hh, (uint32_t*)w_hhB, D3 * DD / 8);
    k_cvt<<<dim3((RR * DD * DD / 8 + 255) / 256), 256, 0, stream>>>(weight, (uint32_t*)weightB, RR * DD * DD / 8);
    k_cvt_adj<<<dim3(RR * TT * NN / 8), 256, 0, stream>>>(adj, (uint32_t*)adjB, rs);

    // FT[t][e][m] = bf16(inputs[m,t,e]);  whwT[e][k] = bf16(whw[k,e])
    k_transpose<<<dim3(16, 16, TT), dim3(32, 8), 0, stream>>>(inputs, (long)DD, TD, FT, (long)DD * NN);
    k_transpose<<<dim3(16, 16, 1), dim3(32, 8), 0, stream>>>(whw, 0L, DD, whwT, 0L);

    k_wcomb<<<dim3(DD / 16, D3 / 16, RR), 32, 0, stream>>>((const uint32_t*)w_ihB,
                                                           (const uint32_t*)weightB, wcombT);
    k_bvec<<<dim3((RR * D3 + 255) / 256), 256, 0, stream>>>(bias, w_ih, bvec);

    k_pmm<<<dim3(DD / 32, NN / 32, RR * TT), 32, 0, stream>>>((const uint32_t*)adjB,
                                                              (const uint32_t*)FT, Pb);
    k_fused<<<dim3(DD / 16, NN / 32, TT), 32, 0, stream>>>(inputs, (const uint32_t*)featsB,
                                                           (const uint32_t*)Pb,
                                                           (const uint32_t*)wcombT,
                                                           (const uint32_t*)whwT,
                                                           (const uint32_t*)w_hhB,
                                                           bvec, rs, b_ih, b_hh, bhw, outp);
}
